// SimpleNetwork_11209864642667
// MI455X (gfx1250) — matrix operand order repaired and checked by reading, not compiled, not run on hardware
//
#include <hip/hip_runtime.h>
#include <hip/hip_bf16.h>
#include <math.h>

// ---------------------------------------------------------------------------
// Equivariant GNN (NequIP-like) for MI455X / gfx1250.
// GEMMs (edge radial MLP, node equivariant linears) on v_wmma_f32_16x16x32_f16.
// Weights pre-swizzled into WMMA B-fragment order -> b128 vector loads.
// LDS staging in A-fragment order -> ds b128 loads. rcp-based SiLU/sigmoid.
// ---------------------------------------------------------------------------

typedef __attribute__((ext_vector_type(16))) _Float16 v16h;
typedef __attribute__((ext_vector_type(8)))  float    v8f;

#define NNODES 50000
#define NEDGES 800000
#define EMB    32
#define HD     64
#define EITER  8            // edge tiles (of 16 edges) per wave
#define INV_SQRT3 0.5773502691896258f

__device__ __forceinline__ float sigm_f(float x) {
  return __builtin_amdgcn_rcpf(1.f + __expf(-x));
}
__device__ __forceinline__ float silu_f(float x) { return x * sigm_f(x); }

__device__ __forceinline__ v8f wmma16(v16h a, v16h b, v8f c) {
  return __builtin_amdgcn_wmma_f32_16x16x32_f16(false, a, false, b, (short)0, c,
                                                false, false);
}

// ---- fragment layout helpers ----------------------------------------------
// A-fragment (16-bit 16x32, ISA 7.12.2): lane = m + 16*(k&8)/8 ; elem =
// (k&7) + 8*(k>=16) within a 32-wide K chunk.  Fragments stored contiguously
// per (chunk, lane) so a consumer does one aligned 32B load.

__device__ __forceinline__ void store_a_frag(_Float16* base, int m, int k, float v) {
  const int ch = k >> 5;
  const int kl = k & 31;
  const int ln = m + ((kl & 8) << 1);        // +16 for hi half
  const int el = (kl & 7) + ((kl & 16) >> 1); // +8 for k>=16
  base[(ch * 32 + ln) * 16 + el] = (_Float16)v;
}

__device__ __forceinline__ v16h load_a_sw(const _Float16* base, int ch, int lane) {
  return *(const v16h*)(base + ((ch * 32 + lane) << 4));
}

// Pre-swizzled B fragments: blob[((kc*NT + nt)*32 + lane)*16 + j]
__device__ __forceinline__ v16h load_b_sw(const _Float16* __restrict__ w,
                                          int NT, int kc, int nt, int lane) {
  return *(const v16h*)(w + ((((kc * NT + nt) << 5) + lane) << 4));
}

// fragment blob sizes: ceil(krows/32) * (ncols/16) * 512 halves
#define FRAG_SZ(KR, NC) ((((KR) + 31) / 32) * ((NC) / 16) * 512)
enum {
  OFF_W1_0 = 0,
  OFF_W2_0 = OFF_W1_0 + FRAG_SZ(64, 64),
  OFF_LS_0 = OFF_W2_0 + FRAG_SZ(64, 64),
  OFF_LV_0 = OFF_LS_0 + FRAG_SZ(32, 64),
  OFF_OS_0 = OFF_LV_0 + FRAG_SZ(32, 32),
  OFF_OV_0 = OFF_OS_0 + FRAG_SZ(64, 32),
  OFF_W1_1 = OFF_OV_0 + FRAG_SZ(32, 16),
  OFF_W2_1 = OFF_W1_1 + FRAG_SZ(64, 64),
  OFF_LS_1 = OFF_W2_1 + FRAG_SZ(64, 96),
  OFF_LV_1 = OFF_LS_1 + FRAG_SZ(48, 96),
  OFF_OS_1 = OFF_LV_1 + FRAG_SZ(48, 48),
  OFF_OV_1 = OFF_OS_1 + FRAG_SZ(80, 32),
  BLOB_HALVES = OFF_OV_1 + FRAG_SZ(64, 16)
};

// ------------------------------- utility kernels ---------------------------

__global__ void zero_kernel(float* __restrict__ p, int n) {
  int i = blockIdx.x * 256 + threadIdx.x;
  if (i < n) p[i] = 0.f;
}

__global__ void cnt_kernel(const int* __restrict__ recv, float* __restrict__ cnt, int e) {
  int i = blockIdx.x * 256 + threadIdx.x;
  if (i < e) atomicAdd(&cnt[recv[i]], 1.f);
}

__global__ void embed_kernel(const int* __restrict__ numbers,
                             const float* __restrict__ embed,
                             float* __restrict__ s, int n) {
  int i = blockIdx.x * 256 + threadIdx.x;
  if (i < n * EMB) {
    int nd = i >> 5, c = i & 31;
    s[i] = embed[numbers[nd] * EMB + c];
  }
}

__global__ void readout_kernel(const float* __restrict__ s,
                               const float* __restrict__ w,
                               float* __restrict__ out, int n) {
  int i = blockIdx.x * 256 + threadIdx.x;
  if (i < n) {
    float acc = 0.f;
#pragma unroll
    for (int c = 0; c < EMB; ++c) acc += s[i * EMB + c] * w[c];
    out[i] = acc;
  }
}

// f32 weights -> f16 fragment-swizzled blob (zero-padded in K)
__device__ void swizzle_b(const float* __restrict__ w, _Float16* __restrict__ dst,
                          int krows, int ncols) {
  const int NT = ncols >> 4;
  const int total = (((krows + 31) >> 5) * NT) << 9;
  for (int i = threadIdx.x; i < total; i += 256) {
    const int el = i & 15;
    const int ln = (i >> 4) & 31;
    const int nt = (i >> 9) % NT;
    const int kc = (i >> 9) / NT;
    const int n  = nt * 16 + (ln & 15);
    const int k  = kc * 32 + ((ln & 16) >> 1) + (el & 7) + ((el & 8) << 1);
    dst[i] = (k < krows) ? (_Float16)w[k * ncols + n] : (_Float16)0.f;
  }
}

__global__ void prep_kernel(const float* w1_0, const float* w2_0,
                            const float* ls0, const float* lv0,
                            const float* os0, const float* ov0,
                            const float* w1_1, const float* w2_1,
                            const float* ls1, const float* lv1,
                            const float* os1, const float* ov1,
                            _Float16* __restrict__ blob) {
  swizzle_b(w1_0, blob + OFF_W1_0, 64, 64);
  swizzle_b(w2_0, blob + OFF_W2_0, 64, 64);
  swizzle_b(ls0,  blob + OFF_LS_0, 32, 64);
  swizzle_b(lv0,  blob + OFF_LV_0, 32, 32);
  swizzle_b(os0,  blob + OFF_OS_0, 64, 32);
  swizzle_b(ov0,  blob + OFF_OV_0, 32, 16);
  swizzle_b(w1_1, blob + OFF_W1_1, 64, 64);
  swizzle_b(w2_1, blob + OFF_W2_1, 64, 96);
  swizzle_b(ls1,  blob + OFF_LS_1, 48, 96);
  swizzle_b(lv1,  blob + OFF_LV_1, 48, 48);
  swizzle_b(os1,  blob + OFF_OS_1, 80, 32);
  swizzle_b(ov1,  blob + OFF_OV_1, 64, 16);
}

// ------------------------------- edge kernel -------------------------------
// One wave32 processes EITER tiles of 16 edges.  GEMM-1 weight fragments are
// hoisted into registers across tiles.  Fused: geometry -> radial MLP (WMMA)
// -> tensor-product message -> scaled atomic scatter.
template <int C0, int C1, int NS, int NI>
__global__ __launch_bounds__(32)
void edge_kernel(const float* __restrict__ pos,
                 const int* __restrict__ senders,
                 const int* __restrict__ receivers,
                 const float* __restrict__ s_cur,
                 const float* __restrict__ v_cur,
                 const float* __restrict__ w0, const float* __restrict__ b0,
                 const float* __restrict__ b1,
                 const _Float16* __restrict__ w1h,
                 const _Float16* __restrict__ w2h,
                 float* __restrict__ agg_s,
                 float* __restrict__ agg_v) {
  constexpr int NV  = NS;
  constexpr int NT2 = NI / 16;
  const int lane = threadIdx.x;

  __shared__ float sh_norm[16];
  __shared__ float sh_rhat[16][3];
  __shared__ int   sh_snd[16];
  __shared__ int   sh_rcv[16];
  __shared__ alignas(32) _Float16 sh_h1f[2 * 32 * 16];  // h1 in A-frag order
  __shared__ float sh_sc[16][NI];

  const int m    = lane & 15;
  const int hb8  = (lane & 16) >> 1;  // C/D row offset for hi half
  const int half = lane >> 4;

  // hoist GEMM-1 (h0 @ w1) B fragments: 2 K-chunks x 4 N-tiles
  v16h b1f[8];
#pragma unroll
  for (int i = 0; i < 8; ++i) b1f[i] = load_b_sw(w1h, 4, i >> 2, i & 3, lane);

  for (int t = 0; t < EITER; ++t) {
    const int ebase = (blockIdx.x * EITER + t) * 16;

    if (lane < 16) {
      const int e  = ebase + lane;
      const int si = senders[e];
      const int ri = receivers[e];
      sh_snd[lane] = si;
      sh_rcv[lane] = ri;
      __builtin_prefetch(s_cur + (size_t)si * C0, 0, 1);  // global_prefetch_b8
      const float dx = pos[ri * 3 + 0] - pos[si * 3 + 0];
      const float dy = pos[ri * 3 + 1] - pos[si * 3 + 1];
      const float dz = pos[ri * 3 + 2] - pos[si * 3 + 2];
      const float nn = sqrtf(dx * dx + dy * dy + dz * dz);
      const float iv = __builtin_amdgcn_rcpf(fmaxf(nn, 1e-9f));
      sh_norm[lane]    = nn;
      sh_rhat[lane][0] = dx * iv;
      sh_rhat[lane][1] = dy * iv;
      sh_rhat[lane][2] = dz * iv;
    }
    __syncthreads();

    // h0 = silu(norm * w0 + b0), built directly in A-fragment layout
    const float nrm = sh_norm[m];
    v16h a0, a1;
#pragma unroll
    for (int j = 0; j < 16; ++j) {
      const int kl = hb8 + (j & 7) + ((j >> 3) << 4);
      a0[j] = (_Float16)silu_f(nrm * w0[kl]      + b0[kl]);
      a1[j] = (_Float16)silu_f(nrm * w0[32 + kl] + b0[32 + kl]);
    }

    // h1 = silu(h0 @ w1 + b1) -> LDS in A-fragment order
#pragma unroll
    for (int nt = 0; nt < 4; ++nt) {
      v8f acc = {};
      acc = wmma16(a0, b1f[nt],     acc);
      acc = wmma16(a1, b1f[4 + nt], acc);
      const float bias = b1[nt * 16 + m];
#pragma unroll
      for (int r = 0; r < 8; ++r)
        store_a_frag(sh_h1f, r + hb8, nt * 16 + m, silu_f(acc[r] + bias));
    }
    __syncthreads();

    // sc = h1 @ w2
    const v16h h1a0 = load_a_sw(sh_h1f, 0, lane);
    const v16h h1a1 = load_a_sw(sh_h1f, 1, lane);
#pragma unroll
    for (int nt = 0; nt < NT2; ++nt) {
      v8f acc = {};
      acc = wmma16(h1a0, load_b_sw(w2h, NT2, 0, nt, lane), acc);
      acc = wmma16(h1a1, load_b_sw(w2h, NT2, 1, nt, lane), acc);
#pragma unroll
      for (int r = 0; r < 8; ++r)
        sh_sc[r + hb8][nt * 16 + m] = acc[r];
    }
    __syncthreads();

    // tensor-product message, scale, atomic scatter
    const int   snd = sh_snd[m];
    const int   rcv = sh_rcv[m];
    const float rh0 = sh_rhat[m][0], rh1 = sh_rhat[m][1], rh2 = sh_rhat[m][2];
    constexpr int HS = NS / 2;

    for (int c = half * HS; c < half * HS + HS; ++c) {   // 0e channels
      float tps;
      if (c < C0) {
        tps = s_cur[(size_t)snd * C0 + c];
      } else {
        const float* vp = &v_cur[((size_t)snd * C1 + (c - C0)) * 3];
        tps = (vp[0] * rh0 + vp[1] * rh1 + vp[2] * rh2) * INV_SQRT3;
      }
      atomicAdd(&agg_s[(size_t)rcv * NS + c], tps * sh_sc[m][c]);
    }
    for (int c = half * HS; c < half * HS + HS; ++c) {   // 1o channels
      float t0, t1, t2;
      if (c < C1) {
        const float* vp = &v_cur[((size_t)snd * C1 + c) * 3];
        t0 = vp[0]; t1 = vp[1]; t2 = vp[2];
      } else {
        const float sv = s_cur[(size_t)snd * C0 + (c - C1)];
        t0 = sv * rh0; t1 = sv * rh1; t2 = sv * rh2;
      }
      const float sc = sh_sc[m][NS + c];
      float* ap = &agg_v[((size_t)rcv * NV + c) * 3];
      atomicAdd(&ap[0], t0 * sc);
      atomicAdd(&ap[1], t1 * sc);
      atomicAdd(&ap[2], t2 * sc);
    }
    __syncthreads();
  }
}

// ------------------------------- node kernel -------------------------------
template <int C0, int C1, int NS, int C0O, int C1O>
__global__ __launch_bounds__(32)
void node_kernel(const float* __restrict__ agg_s,
                 const float* __restrict__ agg_v,
                 const float* __restrict__ cnt,
                 const float* __restrict__ s_cur,
                 const float* __restrict__ v_cur,
                 const _Float16* __restrict__ lin_s_h,
                 const _Float16* __restrict__ lin_v_h,
                 const _Float16* __restrict__ out_s_h,
                 const _Float16* __restrict__ out_v_h,
                 float* __restrict__ s_next,
                 float* __restrict__ v_next) {
  constexpr int NV  = NS;
  constexpr int CS  = NS + C0;           // cat_s K (64 or 80)
  constexpr int CV  = NV + C1;           // cat_v K (32 or 64)
  constexpr int KC1 = (NS + 31) / 32;
  constexpr int KC2 = (CS + 31) / 32;
  constexpr int KC3 = (CV + 31) / 32;

  const int lane  = threadIdx.x;
  const int nbase = blockIdx.x * 16;

  __shared__ alignas(32) _Float16 sh_as[2 * 32 * 16];
  __shared__ alignas(32) _Float16 sh_av[3][2 * 32 * 16];
  __shared__ float                sh_sg[16][NS + NV];
  __shared__ alignas(32) _Float16 sh_cs[3 * 32 * 16];
  __shared__ alignas(32) _Float16 sh_cv[3][2 * 32 * 16];

  for (int i = lane; i < 2 * 512; i += 32)     sh_as[i] = (_Float16)0.f;
  for (int i = lane; i < 3 * 2 * 512; i += 32) (&sh_av[0][0])[i] = (_Float16)0.f;
  for (int i = lane; i < 3 * 512; i += 32)     sh_cs[i] = (_Float16)0.f;
  for (int i = lane; i < 3 * 2 * 512; i += 32) (&sh_cv[0][0])[i] = (_Float16)0.f;

  const int m    = lane & 15;
  const int half = lane >> 4;
  const int hb8  = (lane & 16) >> 1;
  const int node = nbase + m;
  const float inv = __builtin_amdgcn_rcpf(fmaxf(cnt[node], 1.f));

  constexpr int HS = NS / 2;
  for (int c = half * HS; c < half * HS + HS; ++c)
    store_a_frag(sh_as, m, c, agg_s[(size_t)node * NS + c] * inv);
  for (int c = half * HS; c < half * HS + HS; ++c) {
    const float* ap = &agg_v[((size_t)node * NV + c) * 3];
    store_a_frag(sh_av[0], m, c, ap[0] * inv);
    store_a_frag(sh_av[1], m, c, ap[1] * inv);
    store_a_frag(sh_av[2], m, c, ap[2] * inv);
  }
  for (int c = half * (C0 / 2); c < half * (C0 / 2) + C0 / 2; ++c)
    store_a_frag(sh_cs, m, NS + c, s_cur[(size_t)node * C0 + c]);
  if (C1 > 0) {
    for (int c = half * (C1 / 2); c < half * (C1 / 2) + C1 / 2; ++c) {
      const float* vp = &v_cur[((size_t)node * C1 + c) * 3];
      store_a_frag(sh_cv[0], m, NV + c, vp[0]);
      store_a_frag(sh_cv[1], m, NV + c, vp[1]);
      store_a_frag(sh_cv[2], m, NV + c, vp[2]);
    }
  }
  __syncthreads();

  // sg = agg_s @ lin_s  (scalars + gates)
  v16h asf[2];
#pragma unroll
  for (int kc = 0; kc < KC1; ++kc) asf[kc] = load_a_sw(sh_as, kc, lane);
#pragma unroll
  for (int nt = 0; nt < (NS + NV) / 16; ++nt) {
    v8f acc = {};
#pragma unroll
    for (int kc = 0; kc < KC1; ++kc)
      acc = wmma16(asf[kc], load_b_sw(lin_s_h, (NS + NV) / 16, kc, nt, lane), acc);
#pragma unroll
    for (int r = 0; r < 8; ++r)
      sh_sg[r + hb8][nt * 16 + m] = acc[r];
  }
  __syncthreads();

  // new_v = (agg_v @ lin_v) * sigmoid(gates) -> cat_v[:, :NV]
#pragma unroll
  for (int d = 0; d < 3; ++d) {
    v16h avf[2];
#pragma unroll
    for (int kc = 0; kc < KC1; ++kc) avf[kc] = load_a_sw(sh_av[d], kc, lane);
#pragma unroll
    for (int nt = 0; nt < NV / 16; ++nt) {
      v8f acc = {};
#pragma unroll
      for (int kc = 0; kc < KC1; ++kc)
        acc = wmma16(avf[kc], load_b_sw(lin_v_h, NV / 16, kc, nt, lane), acc);
#pragma unroll
      for (int r = 0; r < 8; ++r) {
        const int row = r + hb8, col = nt * 16 + m;
        store_a_frag(sh_cv[d], row, col, acc[r] * sigm_f(sh_sg[row][NS + col]));
      }
    }
  }
  // new_s = silu(scalars) -> cat_s[:, :NS]
  for (int i = lane; i < 16 * NS; i += 32) {
    const int row = i / NS, c = i % NS;
    store_a_frag(sh_cs, row, c, silu_f(sh_sg[row][c]));
  }
  __syncthreads();

  // s_next = cat_s @ out_s
  v16h csf[3];
#pragma unroll
  for (int kc = 0; kc < KC2; ++kc) csf[kc] = load_a_sw(sh_cs, kc, lane);
#pragma unroll
  for (int nt = 0; nt < C0O / 16; ++nt) {
    v8f acc = {};
#pragma unroll
    for (int kc = 0; kc < KC2; ++kc)
      acc = wmma16(csf[kc], load_b_sw(out_s_h, C0O / 16, kc, nt, lane), acc);
#pragma unroll
    for (int r = 0; r < 8; ++r)
      s_next[(size_t)(nbase + r + hb8) * C0O + nt * 16 + m] = acc[r];
  }

  // v_next = cat_v @ out_v  (C1O == 16 -> single N-tile per component)
#pragma unroll
  for (int d = 0; d < 3; ++d) {
    v8f acc = {};
#pragma unroll
    for (int kc = 0; kc < KC3; ++kc)
      acc = wmma16(load_a_sw(sh_cv[d], kc, lane), load_b_sw(out_v_h, 1, kc, 0, lane), acc);
#pragma unroll
    for (int r = 0; r < 8; ++r)
      v_next[((size_t)(nbase + r + hb8) * C1O + m) * 3 + d] = acc[r];
  }
}

// ------------------------------- launcher ----------------------------------

extern "C" void kernel_launch(void* const* d_in, const int* in_sizes, int n_in,
                              void* d_out, int out_size, void* d_ws, size_t ws_size,
                              hipStream_t stream) {
  const float* positions = (const float*)d_in[0];
  const float* p_embed   = (const float*)d_in[1];
  const float* p_lin0_s  = (const float*)d_in[2];
  const float* p_lin0_v  = (const float*)d_in[3];
  const float* p_lin1_s  = (const float*)d_in[4];
  const float* p_lin1_v  = (const float*)d_in[5];
  const float* p_m0_b0   = (const float*)d_in[6];
  const float* p_m0_b1   = (const float*)d_in[7];
  const float* p_m0_w0   = (const float*)d_in[8];
  const float* p_m0_w1   = (const float*)d_in[9];
  const float* p_m0_w2   = (const float*)d_in[10];
  const float* p_m1_b0   = (const float*)d_in[11];
  const float* p_m1_b1   = (const float*)d_in[12];
  const float* p_m1_w0   = (const float*)d_in[13];
  const float* p_m1_w1   = (const float*)d_in[14];
  const float* p_m1_w2   = (const float*)d_in[15];
  const float* p_out0_s  = (const float*)d_in[16];
  const float* p_out0_v  = (const float*)d_in[17];
  const float* p_out1_s  = (const float*)d_in[18];
  const float* p_out1_v  = (const float*)d_in[19];
  const float* p_read_w  = (const float*)d_in[20];
  const int*   numbers   = (const int*)d_in[21];
  const int*   senders   = (const int*)d_in[22];
  const int*   receivers = (const int*)d_in[23];
  float* out = (float*)d_out;
  (void)in_sizes; (void)n_in; (void)out_size; (void)ws_size;

  float* wsf = (float*)d_ws;
  size_t o = 0;
  float* cnt   = wsf + o; o += NNODES;
  float* agg_s = wsf + o; o += (size_t)NNODES * 48;
  float* agg_v = wsf + o; o += (size_t)NNODES * 48 * 3;
  float* sA    = wsf + o; o += (size_t)NNODES * 32;
  float* sB    = wsf + o; o += (size_t)NNODES * 32;
  float* vA    = wsf + o; o += (size_t)NNODES * 16 * 3;
  float* vB    = wsf + o; o += (size_t)NNODES * 16 * 3;
  o = (o + 7) & ~(size_t)7;                 // 32B-align the f16 blob
  _Float16* blob = (_Float16*)(wsf + o);

  const int Z = 256;
  auto blocks = [](size_t n, int b) { return (unsigned)((n + b - 1) / b); };

  prep_kernel<<<1, Z, 0, stream>>>(p_m0_w1, p_m0_w2, p_lin0_s, p_lin0_v,
                                   p_out0_s, p_out0_v,
                                   p_m1_w1, p_m1_w2, p_lin1_s, p_lin1_v,
                                   p_out1_s, p_out1_v, blob);

  zero_kernel<<<blocks(NNODES, Z), Z, 0, stream>>>(cnt, NNODES);
  cnt_kernel<<<blocks(NEDGES, Z), Z, 0, stream>>>(receivers, cnt, NEDGES);
  embed_kernel<<<blocks((size_t)NNODES * EMB, Z), Z, 0, stream>>>(numbers, p_embed, sA, NNODES);

  // ---- layer 0: C0=32, C1=0, NS=NV=32, NI=64 ----
  zero_kernel<<<blocks((size_t)NNODES * 32, Z), Z, 0, stream>>>(agg_s, NNODES * 32);
  zero_kernel<<<blocks((size_t)NNODES * 96, Z), Z, 0, stream>>>(agg_v, NNODES * 96);
  edge_kernel<32, 0, 32, 64><<<NEDGES / (16 * EITER), 32, 0, stream>>>(
      positions, senders, receivers, sA, sA /*unused*/,
      p_m0_w0, p_m0_b0, p_m0_b1,
      blob + OFF_W1_0, blob + OFF_W2_0, agg_s, agg_v);
  node_kernel<32, 0, 32, 32, 16><<<NNODES / 16, 32, 0, stream>>>(
      agg_s, agg_v, cnt, sA, vA /*unused*/,
      blob + OFF_LS_0, blob + OFF_LV_0, blob + OFF_OS_0, blob + OFF_OV_0,
      sB, vB);

  // ---- layer 1: C0=32, C1=16, NS=NV=48, NI=96 ----
  zero_kernel<<<blocks((size_t)NNODES * 48, Z), Z, 0, stream>>>(agg_s, NNODES * 48);
  zero_kernel<<<blocks((size_t)NNODES * 144, Z), Z, 0, stream>>>(agg_v, NNODES * 144);
  edge_kernel<32, 16, 48, 96><<<NEDGES / (16 * EITER), 32, 0, stream>>>(
      positions, senders, receivers, sB, vB,
      p_m1_w0, p_m1_b0, p_m1_b1,
      blob + OFF_W1_1, blob + OFF_W2_1, agg_s, agg_v);
  node_kernel<32, 16, 48, 32, 16><<<NNODES / 16, 32, 0, stream>>>(
      agg_s, agg_v, cnt, sB, vB,
      blob + OFF_LS_1, blob + OFF_LV_1, blob + OFF_OS_1, blob + OFF_OV_1,
      sA, vA);

  readout_kernel<<<blocks(NNODES, Z), Z, 0, stream>>>(sA, p_read_w, out, NNODES);
}
